// S5Block_26903675142795
// MI455X (gfx1250) — matrix operand order repaired and checked by reading, not compiled, not run on hardware
//
#include <hip/hip_runtime.h>

typedef __bf16 bf16;
typedef bf16  v16bf __attribute__((ext_vector_type(16)));
typedef bf16  v8bf  __attribute__((ext_vector_type(8)));
typedef float v8f   __attribute__((ext_vector_type(8)));

#define DIM      1024
#define HEADS    8
#define HEAD_DIM 128
#define D_STATE  256
#define BATCH    8192
#define BT       32    // batch tile (2 x 16-wide WMMA column tiles)
#define XS       136   // padded x-tile row stride (bf16 elems), bank-conflict free
#define NSS      264   // padded ns row stride (bf16 elems), bank-conflict free

// Fragment buffer sizes (bf16 elements)
#define BBAR_FRAG_ELEMS (HEADS * 16 * 4 * 32 * 16)   // 262144 per (re/im)
#define C_FRAG_ELEMS    (HEADS * 8 * 8 * 32 * 16)    // 262144 per (re/imneg)

__device__ __forceinline__ v8f wmma_bf16(v16bf a, v16bf b, v8f c) {
  return __builtin_amdgcn_wmma_f32_16x16x32_bf16(false, a, false, b,
                                                 (short)0, c, false, false);
}

// Load a 16-element bf16 fragment from LDS as two 16B chunks (16B-aligned).
__device__ __forceinline__ v16bf load16_lds(const bf16* p) {
  const v8bf* q = (const v8bf*)p;
  v8bf lo = q[0];
  v8bf hi = q[1];
  v16bf r;
#pragma unroll
  for (int i = 0; i < 8; ++i) { r[i] = lo[i]; r[i + 8] = hi[i]; }
  return r;
}

// ---------------------------------------------------------------------------
// Precompute 1: A_bar = exp(softplus(log_dt) * A),  A = (-exp(A_re_log), A_im)
// ---------------------------------------------------------------------------
__global__ void k_abar(const float* __restrict__ A_re_log,
                       const float* __restrict__ A_im,
                       const float* __restrict__ log_dt,
                       float* __restrict__ abr, float* __restrict__ abi) {
  int idx = blockIdx.x * blockDim.x + threadIdx.x;   // < HEADS*D_STATE
  float ld  = log_dt[idx];
  float dt  = (ld > 20.0f) ? ld : log1pf(expf(ld));  // softplus
  float are = -expf(A_re_log[idx]);
  float aim = A_im[idx];
  float e   = expf(dt * are);
  abr[idx] = e * cosf(dt * aim);
  abi[idx] = e * sinf(dt * aim);
}

// ---------------------------------------------------------------------------
// Precompute 2: B_bar = ((A_bar - 1)/A) * B  -> bf16, WMMA A-fragment order.
// A-fragment (16x32 bf16): element(lane, i) = W[M=16*mt + (lane&15)]
//                                              [K=32*kt + (lane>>4)*16 + i]
// ---------------------------------------------------------------------------
__global__ void k_bbar(const float* __restrict__ B_re, const float* __restrict__ B_im,
                       const float* __restrict__ A_re_log, const float* __restrict__ A_im,
                       const float* __restrict__ abr, const float* __restrict__ abi,
                       bf16* __restrict__ fr, bf16* __restrict__ fi) {
  int idx = blockIdx.x * blockDim.x + threadIdx.x;   // h*N*P + n*P + p
  int p = idx & (HEAD_DIM - 1);
  int n = (idx >> 7) & (D_STATE - 1);
  int h = idx >> 15;
  int hn = h * D_STATE + n;

  float are = -expf(A_re_log[hn]);
  float aim = A_im[hn];
  float nre = abr[hn] - 1.0f, nim = abi[hn];
  float inv = 1.0f / (are * are + aim * aim);
  float cr  = (nre * are + nim * aim) * inv;         // (A_bar-1)/A
  float ci  = (nim * are - nre * aim) * inv;
  float br = B_re[idx], bi = B_im[idx];
  float vr = cr * br - ci * bi;
  float vi = cr * bi + ci * br;

  int mt = n >> 4, laneLo = n & 15;
  int kt = p >> 5, rem = p & 31;
  int hi = rem >> 4;
  int i  = rem & 15;
  int lane = laneLo + (hi << 4);
  size_t fo = ((((size_t)(h * 16 + mt) * 4 + kt) * 32 + lane) << 4) + i;
  fr[fo] = (bf16)vr;
  fi[fo] = (bf16)vi;
}

// ---------------------------------------------------------------------------
// Precompute 3: C_re and -C_im -> bf16 A-fragment order (M=p tiles, K=n).
// ---------------------------------------------------------------------------
__global__ void k_cfrag(const float* __restrict__ C_re, const float* __restrict__ C_im,
                        bf16* __restrict__ fre, bf16* __restrict__ fimn) {
  int idx = blockIdx.x * blockDim.x + threadIdx.x;   // h*P*N + p*N + n
  int n = idx & (D_STATE - 1);
  int p = (idx >> 8) & (HEAD_DIM - 1);
  int h = idx >> 15;

  int pt = p >> 4, laneLo = p & 15;
  int kt = n >> 5, rem = n & 31;
  int hi = rem >> 4;
  int i  = rem & 15;
  int lane = laneLo + (hi << 4);
  size_t fo = ((((size_t)(h * 8 + pt) * 8 + kt) * 32 + lane) << 4) + i;
  fre[fo]  = (bf16)C_re[idx];
  fimn[fo] = (bf16)(-C_im[idx]);
}

// ---------------------------------------------------------------------------
// RMSNorm scale: one wave (32 lanes) per row.
// ---------------------------------------------------------------------------
__global__ void k_rms(const float* __restrict__ x, float* __restrict__ scale) {
  int row  = blockIdx.x * 4 + (threadIdx.x >> 5);
  int lane = threadIdx.x & 31;
  const float4* xr = (const float4*)(x + (size_t)row * DIM);
  float s = 0.0f;
#pragma unroll
  for (int j = 0; j < 8; ++j) {
    float4 v = xr[j * 32 + lane];
    s += v.x * v.x + v.y * v.y + v.z * v.z + v.w * v.w;
  }
#pragma unroll
  for (int o = 16; o > 0; o >>= 1) s += __shfl_xor(s, o, 32);
  if (lane == 0) scale[row] = rsqrtf(s * (1.0f / (float)DIM) + 1e-4f);
}

// ---------------------------------------------------------------------------
// Main fused kernel: per (batch-tile of 32, head).
// 8 waves: GEMM1 (Bx) -> complex recurrence -> GEMM2 (Ch.real) -> epilogue.
// Each weight A-fragment feeds two batch-column WMMAs (2x reuse).
// ---------------------------------------------------------------------------
__global__ __launch_bounds__(256) void k_main(
    const float* __restrict__ x_t,
    const float* __restrict__ st_re, const float* __restrict__ st_im,
    const float* __restrict__ norm_w, const float* __restrict__ Dp,
    const float* __restrict__ scale,
    const float* __restrict__ abr, const float* __restrict__ abi,
    const bf16* __restrict__ bbar_re, const bf16* __restrict__ bbar_im,
    const bf16* __restrict__ c_re, const bf16* __restrict__ c_imn,
    float* __restrict__ out,
    float* __restrict__ ns_re_out, float* __restrict__ ns_im_out) {
  __shared__ bf16 xs[BT * XS];
  __shared__ bf16 nsr[BT * NSS];
  __shared__ bf16 nsi[BT * NSS];

  const int h    = blockIdx.y;
  const int b0   = blockIdx.x * BT;
  const int tid  = threadIdx.x;
  const int lane = tid & 31;
  const int w    = tid >> 5;          // wave id 0..7
  const int bcol = lane & 15;         // WMMA N index within a column tile
  const int hiw  = lane >> 4;         // half-wave

  const bf16* crbase = c_re  + ((size_t)(h * 8 + w) * 8) * 512;
  const bf16* cnbase = c_imn + ((size_t)(h * 8 + w) * 8) * 512;
  // Hint: GEMM2 weights will be needed soon (global_prefetch path).
  __builtin_prefetch((const void*)(crbase + (size_t)lane * 16), 0, 3);
  __builtin_prefetch((const void*)(cnbase + (size_t)lane * 16), 0, 3);

  // ---- Stage normalized x tile (bf16) into LDS: 32 rows x 128 cols ----
  {
    int r  = tid >> 3;                // 0..31 rows
    int c0 = (tid & 7) * 16;          // 0..112
    const float* xp = x_t + (size_t)(b0 + r) * DIM + h * HEAD_DIM + c0;
    float sc = scale[b0 + r];
    v8f xv0 = *(const v8f*)xp;
    v8f xv1 = *(const v8f*)(xp + 8);
    v8f wv0 = *(const v8f*)(norm_w + h * HEAD_DIM + c0);
    v8f wv1 = *(const v8f*)(norm_w + h * HEAD_DIM + c0 + 8);
    v8bf t0, t1;
#pragma unroll
    for (int i = 0; i < 8; ++i) {
      t0[i] = (bf16)(xv0[i] * sc * wv0[i]);
      t1[i] = (bf16)(xv1[i] * sc * wv1[i]);
    }
    *(v8bf*)&xs[r * XS + c0]     = t0;
    *(v8bf*)&xs[r * XS + c0 + 8] = t1;
  }
  __syncthreads();

  // ---- GEMM1: Bx[n,b] = sum_p Bbar[n,p] * x[b,p]  (re & im), fused recurrence ----
#pragma unroll
  for (int mi = 0; mi < 2; ++mi) {
    const int mt = w + mi * 8;        // n-tile 0..15
    v8f accr[2], acci[2];
#pragma unroll
    for (int c = 0; c < 2; ++c) {
      accr[c] = (v8f){0.f, 0.f, 0.f, 0.f, 0.f, 0.f, 0.f, 0.f};
      acci[c] = (v8f){0.f, 0.f, 0.f, 0.f, 0.f, 0.f, 0.f, 0.f};
    }
    const bf16* frbase = bbar_re + ((size_t)(h * 16 + mt) * 4) * 512;
    const bf16* fibase = bbar_im + ((size_t)(h * 16 + mt) * 4) * 512;
#pragma unroll
    for (int kt = 0; kt < 4; ++kt) {
      v16bf ar = *(const v16bf*)(frbase + ((size_t)kt * 32 + lane) * 16);
      v16bf ai = *(const v16bf*)(fibase + ((size_t)kt * 32 + lane) * 16);
      v16bf bx0 = load16_lds(&xs[bcol * XS + kt * 32 + hiw * 16]);
      v16bf bx1 = load16_lds(&xs[(16 + bcol) * XS + kt * 32 + hiw * 16]);
      accr[0] = wmma_bf16(ar, bx0, accr[0]);
      acci[0] = wmma_bf16(ai, bx0, acci[0]);
      accr[1] = wmma_bf16(ar, bx1, accr[1]);
      acci[1] = wmma_bf16(ai, bx1, acci[1]);
    }
    // Complex recurrence: ns = A_bar * state + Bx  (D-fragment lanes -> state rows)
    const int n0 = mt * 16 + hiw * 8;
    v8f ab_r = *(const v8f*)(abr + h * D_STATE + n0);
    v8f ab_i = *(const v8f*)(abi + h * D_STATE + n0);
#pragma unroll
    for (int c = 0; c < 2; ++c) {
      const int brow = c * 16 + bcol;
      const size_t sbase = ((size_t)(b0 + brow) * HEADS + h) * D_STATE + n0;
      v8f sr = *(const v8f*)(st_re + sbase);
      v8f si = *(const v8f*)(st_im + sbase);
      v8f nr = ab_r * sr - ab_i * si + accr[c];
      v8f ni = ab_r * si + ab_i * sr + acci[c];
      *(v8f*)(ns_re_out + sbase) = nr;
      *(v8f*)(ns_im_out + sbase) = ni;
      v8bf pr, pi;
#pragma unroll
      for (int i = 0; i < 8; ++i) { pr[i] = (bf16)nr[i]; pi[i] = (bf16)ni[i]; }
      *(v8bf*)&nsr[brow * NSS + n0] = pr;
      *(v8bf*)&nsi[brow * NSS + n0] = pi;
    }
  }
  __syncthreads();

  // ---- GEMM2: Ch_re[p,b] = C_re.ns_re + (-C_im).ns_im ----
  v8f acc[2];
  acc[0] = (v8f){0.f, 0.f, 0.f, 0.f, 0.f, 0.f, 0.f, 0.f};
  acc[1] = (v8f){0.f, 0.f, 0.f, 0.f, 0.f, 0.f, 0.f, 0.f};
#pragma unroll
  for (int kt = 0; kt < 8; ++kt) {
    v16bf a1 = *(const v16bf*)(crbase + ((size_t)kt * 32 + lane) * 16);
    v16bf b1c0 = load16_lds(&nsr[bcol * NSS + kt * 32 + hiw * 16]);
    v16bf b1c1 = load16_lds(&nsr[(16 + bcol) * NSS + kt * 32 + hiw * 16]);
    acc[0] = wmma_bf16(a1, b1c0, acc[0]);
    acc[1] = wmma_bf16(a1, b1c1, acc[1]);
    v16bf a2 = *(const v16bf*)(cnbase + ((size_t)kt * 32 + lane) * 16);
    v16bf b2c0 = load16_lds(&nsi[bcol * NSS + kt * 32 + hiw * 16]);
    v16bf b2c1 = load16_lds(&nsi[(16 + bcol) * NSS + kt * 32 + hiw * 16]);
    acc[0] = wmma_bf16(a2, b2c0, acc[0]);
    acc[1] = wmma_bf16(a2, b2c1, acc[1]);
  }

  // ---- Epilogue: out = x_t + 2*Ch_re + D * x_norm ----
  {
    const int p0  = w * 16 + hiw * 8;
    const int col = h * HEAD_DIM + p0;
    v8f wv = *(const v8f*)(norm_w + col);
    v8f dv = *(const v8f*)(Dp + col);
#pragma unroll
    for (int c = 0; c < 2; ++c) {
      const int brow = c * 16 + bcol;
      const size_t ro = (size_t)(b0 + brow) * DIM + col;
      float sc = scale[b0 + brow];
      v8f xv = *(const v8f*)(x_t + ro);
      v8f o;
#pragma unroll
      for (int i = 0; i < 8; ++i) {
        float xn = xv[i] * sc * wv[i];
        o[i] = xv[i] + 2.0f * acc[c][i] + dv[i] * xn;
      }
      *(v8f*)(out + ro) = o;
    }
  }
}

// ---------------------------------------------------------------------------
extern "C" void kernel_launch(void* const* d_in, const int* in_sizes, int n_in,
                              void* d_out, int out_size, void* d_ws, size_t ws_size,
                              hipStream_t stream) {
  const float* x_t      = (const float*)d_in[0];
  const float* state_re = (const float*)d_in[1];
  const float* state_im = (const float*)d_in[2];
  const float* norm_w   = (const float*)d_in[3];
  const float* A_re_log = (const float*)d_in[4];
  const float* A_im     = (const float*)d_in[5];
  const float* B_re     = (const float*)d_in[6];
  const float* B_im     = (const float*)d_in[7];
  const float* C_re     = (const float*)d_in[8];
  const float* C_im     = (const float*)d_in[9];
  const float* Dp       = (const float*)d_in[10];
  const float* log_dt   = (const float*)d_in[11];

  float* out   = (float*)d_out;
  float* ns_re = out + (size_t)BATCH * DIM;
  float* ns_im = ns_re + (size_t)BATCH * HEADS * D_STATE;

  // Workspace layout
  float* abr   = (float*)d_ws;
  float* abi   = abr + HEADS * D_STATE;
  float* scale = abi + HEADS * D_STATE;
  bf16*  bbr   = (bf16*)(scale + BATCH);
  bf16*  bbi   = bbr + BBAR_FRAG_ELEMS;
  bf16*  cre   = bbi + BBAR_FRAG_ELEMS;
  bf16*  cin   = cre + C_FRAG_ELEMS;

  k_abar<<<(HEADS * D_STATE) / 256, 256, 0, stream>>>(A_re_log, A_im, log_dt, abr, abi);
  k_bbar<<<(HEADS * D_STATE * HEAD_DIM) / 256, 256, 0, stream>>>(
      B_re, B_im, A_re_log, A_im, abr, abi, bbr, bbi);
  k_cfrag<<<(HEADS * HEAD_DIM * D_STATE) / 256, 256, 0, stream>>>(C_re, C_im, cre, cin);
  k_rms<<<BATCH / 4, 128, 0, stream>>>(x_t, scale);
  k_main<<<dim3(BATCH / BT, HEADS), 256, 0, stream>>>(
      x_t, state_re, state_im, norm_w, Dp, scale, abr, abi,
      bbr, bbi, cre, cin, out, ns_re, ns_im);
}
